// SEGNNModel_24189255811501
// MI455X (gfx1250) — matrix operand order repaired and checked by reading, not compiled, not run on hardware
//
// SEGNN forward for MI455X (gfx1250, wave32, WMMA).
// HBM-bound problem -> fp32 state in memory; all equivariant linear layers
// densified (block-diagonal * 1/sqrt(fan)) into f16 matrices PRE-PACKED in
// WMMA-fragment order, executed with v_wmma_f32_16x16x32_f16 (f32 accum),
// 4 N-tiles per wave for A-fragment reuse. Irregular edge tensor-product is
// an LDS-staged VALU kernel. Wigner-3j built on-device (fp64): capture-safe.
#include <hip/hip_runtime.h>
#include <math.h>

typedef _Float16 h16;
typedef __attribute__((ext_vector_type(16))) _Float16 v16h;
typedef __attribute__((ext_vector_type(8)))  _Float16 v8h;
typedef __attribute__((ext_vector_type(8)))  float    v8f;

#define EDGE_CHUNK 20000   // divides 320000; multiple of 16

// ---------------- Wigner-3j tables (15 tensors, 615 floats) ----------------
__device__ const int d_L1[15]  = {0,1,2,0,1,1,1,2,2,0,2,1,1,2,2};
__device__ const int d_L2[15]  = {0,1,2,1,0,1,2,1,2,2,0,1,2,1,2};
__device__ const int d_L3[15]  = {0,0,0,1,1,1,1,1,1,2,2,2,2,2,2};
__device__ const int d_OFF[15] = {0,1,10,35,44,53,80,125,170,245,270,295,340,415,490};
static const int H_L1[15]  = {0,1,2,0,1,1,1,2,2,0,2,1,1,2,2};
static const int H_L2[15]  = {0,1,2,1,0,1,2,1,2,2,0,1,2,1,2};
static const int H_L3[15]  = {0,0,0,1,1,1,1,1,1,2,2,2,2,2,2};
static const int H_OFF[15] = {0,1,10,35,44,53,80,125,170,245,270,295,340,415,490};

struct dc { double re, im; };
__device__ inline dc cmul(dc a, dc b){ dc r; r.re=a.re*b.re-a.im*b.im; r.im=a.re*b.im+a.im*b.re; return r; }
__device__ inline dc cconj(dc a){ dc r; r.re=a.re; r.im=-a.im; return r; }

__device__ double d_fact(int n){ double r=1.0; for(int i=2;i<=n;++i) r*=(double)i; return r; }

__device__ double d_cg(int j1,int m1,int j2,int m2,int j3,int m3){
  if (m1+m2!=m3) return 0.0;
  double pre = sqrt((2.0*j3+1.0)*d_fact(j1+j2-j3)*d_fact(j1-j2+j3)*d_fact(-j1+j2+j3)/d_fact(j1+j2+j3+1));
  pre *= sqrt(d_fact(j1+m1)*d_fact(j1-m1)*d_fact(j2+m2)*d_fact(j2-m2)*d_fact(j3+m3)*d_fact(j3-m3));
  double s=0.0;
  for (int k=0;k<=j1+j2-j3;++k){
    int d0=k,d1=j1+j2-j3-k,d2=j1-m1-k,d3=j2+m2-k,d4=j3-j2+m1+k,d5=j3-j1-m2+k;
    if (d0<0||d1<0||d2<0||d3<0||d4<0||d5<0) continue;
    double den=d_fact(d0)*d_fact(d1)*d_fact(d2)*d_fact(d3)*d_fact(d4)*d_fact(d5);
    s += ((k&1)? -1.0:1.0)/den;
  }
  return pre*s;
}

__device__ void d_rbasis(int l, dc A[5][5]){
  for (int a=0;a<5;++a) for (int b=0;b<5;++b){ A[a][b].re=0.0; A[a][b].im=0.0; }
  A[l][l].re = 1.0;
  const double is2 = 0.70710678118654752440;
  for (int m=1;m<=l;++m){
    double sg = (m&1)? -1.0 : 1.0;
    A[l+m][l+m].re =  sg*is2;
    A[l+m][l-m].re =  is2;
    A[l-m][l-m].im =  is2;
    A[l-m][l+m].im = -sg*is2;
  }
}

__global__ void k_init_w3j(float* out){
  int t = threadIdx.x;
  if (t >= 15) return;
  int l1=d_L1[t], l2=d_L2[t], l3=d_L3[t], off=d_OFF[t];
  int n1=2*l1+1, n2=2*l2+1, n3=2*l3+1;
  double C[5][5][5];
  for (int a=0;a<5;++a) for (int b=0;b<5;++b) for (int c=0;c<5;++c) C[a][b][c]=0.0;
  for (int m1=-l1;m1<=l1;++m1) for (int m2=-l2;m2<=l2;++m2){
    int m3=m1+m2;
    if (m3>=-l3 && m3<=l3) C[l1+m1][l2+m2][l3+m3]=d_cg(l1,m1,l2,m2,l3,m3);
  }
  dc A1[5][5],A2[5][5],A3[5][5];
  d_rbasis(l1,A1); d_rbasis(l2,A2); d_rbasis(l3,A3);
  double Rr[125], Ri[125];
  double mr=0.0, mi=0.0;
  for (int a=0;a<n1;++a) for (int b=0;b<n2;++b) for (int c=0;c<n3;++c){
    dc s; s.re=0.0; s.im=0.0;
    for (int m=0;m<n1;++m) for (int n=0;n<n2;++n) for (int p=0;p<n3;++p){
      double cv=C[m][n][p];
      if (cv==0.0) continue;
      dc q = cmul(cmul(cconj(A1[a][m]), cconj(A2[b][n])), A3[c][p]);
      s.re += q.re*cv; s.im += q.im*cv;
    }
    int idx=(a*n2+b)*n3+c;
    Rr[idx]=s.re; Ri[idx]=s.im;
    if (fabs(s.re)>mr) mr=fabs(s.re);
    if (fabs(s.im)>mi) mi=fabs(s.im);
  }
  const double* R = (mi>mr)? Ri : Rr;
  int tot=n1*n2*n3;
  double nrm=0.0; for (int i=0;i<tot;++i) nrm+=R[i]*R[i];
  nrm = sqrt(nrm);
  for (int i=0;i<tot;++i) out[off+i]=(float)(R[i]/nrm);
}

// ---------------- gate (gi(310) -> HID(256)) ----------------
__device__ __forceinline__ float sigf(float x){ return 1.f/(1.f+__expf(-x)); }
__device__ __forceinline__ float gate_val(const float* g, int c){
  if (c < 54){ float s=g[c]; return s*sigf(s); }
  if (c < 105){ int u=(c-54)/3;  return g[108+(c-54)]  * sigf(g[54+u]); }
  if (c < 156){ int u=(c-105)/3; return g[159+(c-105)] * sigf(g[71+u]); }
  if (c < 206){ int u=(c-156)/5; return g[210+(c-156)] * sigf(g[88+u]); }
  { int u=(c-206)/5; return g[260+(c-206)] * sigf(g[98+u]); }
}

// ---------------- message tensor-product kernel ----------------
#define MSG_MAXP 60
struct MsgDesc {
  const float* w[MSG_MAXP];
  int   in_off[MSG_MAXP];
  int   t_off[MSG_MAXP];
  short cg[MSG_MAXP];
  char  m1v[MSG_MAXP], m3v[MSG_MAXP], n1v[MSG_MAXP], n2v[MSG_MAXP], n3v[MSG_MAXP], kk[MSG_MAXP];
  float kscale[6];
  int npaths, din, ttot;
};
__device__ const int d_KOFF[7] = {0,54,108,159,210,260,310};
__device__ const int d_KN3[6]  = {1,1,3,3,5,5};

__global__ void __launch_bounds__(64)
k_msg(MsgDesc D, const float* x, const float* pos, const int* ei,
      const float* cell, const float* w3j, const float* bias,
      h16* Mh, int e0, int E){
  int e = e0 + blockIdx.x;
  if (e >= E) return;
  int tid = threadIdx.x;
  __shared__ float s_x[513];
  __shared__ float s_Y[9];
  __shared__ float s_T[3340];
  __shared__ float s_g[310];
  int src = ei[e], dst = ei[E + e];
  int din = D.din;
  if (tid == 0){
    float rx = pos[dst*3+0]-pos[src*3+0]+cell[(size_t)e*3+0];
    float ry = pos[dst*3+1]-pos[src*3+1]+cell[(size_t)e*3+1];
    float rz = pos[dst*3+2]-pos[src*3+2]+cell[(size_t)e*3+2];
    const float s3 = 1.7320508075688772f, s5 = 2.2360679774997896f;
    s_Y[0] = 1.f;
    s_Y[1] = s3*rx; s_Y[2] = s3*ry; s_Y[3] = s3*rz;
    s_Y[4] = s5*(s3*rx*rz);
    s_Y[5] = s5*(s3*rx*ry);
    s_Y[6] = s5*(ry*ry - 0.5f*(rx*rx+rz*rz));
    s_Y[7] = s5*(s3*ry*rz);
    s_Y[8] = s5*(0.5f*s3*(rz*rz-rx*rx));
    s_x[2*din] = sqrtf(rx*rx+ry*ry+rz*rz);
  }
  __builtin_prefetch(x + (size_t)src*din, 0, 0);
  for (int c = tid; c < din; c += blockDim.x){
    s_x[c]       = x[(size_t)dst*din + c];
    s_x[din + c] = x[(size_t)src*din + c];
  }
  __syncthreads();
  // Stage 1: T[path][u,p] = sum_{m,n} x[u,m] * Y[n] * C[m,n,p]
  for (int p = 0; p < D.npaths; ++p){
    int n1 = D.n1v[p], n2 = D.n2v[p], n3 = D.n3v[p];
    int m1 = D.m1v[p];
    int cnt = m1*n3;
    int io = D.in_off[p], to = D.t_off[p], cg = D.cg[p];
    int yb = (n2==1)?0:((n2==3)?1:4);
    for (int idx = tid; idx < cnt; idx += blockDim.x){
      int u = idx / n3, pp = idx % n3;
      float acc = 0.f;
      for (int m = 0; m < n1; ++m){
        float xv = s_x[io + u*n1 + m];
        for (int n = 0; n < n2; ++n)
          acc += xv * s_Y[yb+n] * w3j[cg + (m*n2+n)*n3 + pp];
      }
      s_T[to + idx] = acc;
    }
  }
  __syncthreads();
  // Stage 2: g[c] = scale_k * sum_paths sum_u W[u,w]*T[u,p]  (+bias on k0)
  for (int c = tid; c < 310; c += blockDim.x){
    int k = 0;
    while (c >= d_KOFF[k+1]) ++k;
    int n3 = d_KN3[k];
    int rel = c - d_KOFF[k];
    int wv = rel / n3, pp = rel % n3;
    float acc = 0.f;
    for (int p = 0; p < D.npaths; ++p){
      if (D.kk[p] != (char)k) continue;
      const float* W = D.w[p];
      int m1 = D.m1v[p], m3 = D.m3v[p], to = D.t_off[p];
      for (int u = 0; u < m1; ++u)
        acc += W[u*m3 + wv] * s_T[to + u*n3 + pp];
    }
    acc *= D.kscale[k];
    if (c < 54) acc += bias[c];
    s_g[c] = acc;
  }
  __syncthreads();
  for (int c = tid; c < 256; c += blockDim.x)
    Mh[(size_t)(e - e0)*256 + c] = (h16)gate_val(s_g, c);
}

// ---------------- dense-weight build, packed into WMMA fragment order -----
// Bp layout: tile (kt = k/32, nt = n/16) stored as 512 halves:
//   [lane(32)][t(16)] with lane = hf*16 + (n%16), element t = (k%32)&15,
//   hf = (k%32)>>4.  GEMM loads one contiguous 32B v16h per lane per tile.
#define LIN_MAXP 12
struct LinDesc {
  const float* w[LIN_MAXP];
  int in_off[LIN_MAXP], out_off[LIN_MAXP];
  int m1v[LIN_MAXP], m3v[LIN_MAXP], n3v[LIN_MAXP], ebase[LIN_MAXP];
  float scale[LIN_MAXP];
  int npaths, etotal, Kpad, Npad;
};

__device__ __forceinline__ size_t pack_idx(int r, int c, int ntiles){
  int kt = r >> 5, rr = r & 31;
  int nt = c >> 4, cc = c & 15;
  return (((size_t)kt*ntiles + nt)*32 + (rr >> 4)*16 + cc)*16 + (rr & 15);
}

__global__ void k_densify(LinDesc D, h16* Bp){
  int idx = blockIdx.x*blockDim.x + threadIdx.x;
  if (idx >= D.etotal) return;
  int p = 0;
  while (p+1 < D.npaths && idx >= D.ebase[p+1]) ++p;
  int r = idx - D.ebase[p];
  int m3 = D.m3v[p], n3 = D.n3v[p];
  int u = r/(m3*n3); int r2 = r%(m3*n3); int wv = r2/n3; int pp = r2%n3;
  float v = D.w[p][u*m3 + wv] * D.scale[p];
  int row = D.in_off[p] + u*n3 + pp;
  int col = D.out_off[p] + wv*n3 + pp;
  Bp[pack_idx(row, col, D.Npad >> 4)] = (h16)v;
}

// ---------------- WMMA GEMM ------------------------------------------------
// C[MxN] = A[MxK](f16,row-major) * Bpacked[KxN](f16) + bias(cols<nb_)
// One wave computes a 16x64 tile (1 M-tile x 4 N-tiles): per 32-K step,
// 2 x v8h A-loads are reused by 4 WMMAs fed by contiguous v16h B-loads.
__global__ void __launch_bounds__(256)
k_gemm(const h16* __restrict__ A, const h16* __restrict__ Bp,
       const float* __restrict__ bias, int nb_,
       float* __restrict__ C, int M, int N, int K){
  __builtin_amdgcn_s_wait_tensorcnt(0);
  int wid = blockIdx.x*(blockDim.x>>5) + (threadIdx.x>>5);
  int ntiles = N >> 4;
  int ngroups = ntiles >> 2;          // N-tiles processed 4 at a time
  int mtiles = M >> 4;
  if (wid >= mtiles*ngroups) return;
  int m0 = (wid / ngroups) << 4;
  int nt0 = (wid % ngroups) << 2;     // first n-tile index
  int lane = threadIdx.x & 31;
  int hf = lane >> 4, l16 = lane & 15;
  v8f acc[4] = {};
  const h16* arow = A + (size_t)(m0 + l16)*K + hf*16;
  for (int k0 = 0; k0 < K; k0 += 32){
    v8h alo = *(const v8h*)(arow + k0);
    v8h ahi = *(const v8h*)(arow + k0 + 8);
    v16h a;
#pragma unroll
    for (int t = 0; t < 8; ++t){ a[t] = alo[t]; a[8+t] = ahi[t]; }
    const h16* bbase = Bp + (((size_t)(k0 >> 5)*ntiles + nt0)*32 + lane)*16;
#pragma unroll
    for (int q = 0; q < 4; ++q){
      v16h b = *(const v16h*)(bbase + q*512);
      acc[q] = __builtin_amdgcn_wmma_f32_16x16x32_f16(false, a, false, b, (short)0, acc[q], false, false);
    }
  }
#pragma unroll
  for (int q = 0; q < 4; ++q){
    int col = (nt0 + q)*16 + l16;
    float bv = (col < nb_) ? bias[col] : 0.f;
#pragma unroll
    for (int r = 0; r < 8; ++r)
      C[(size_t)(m0 + hf*8 + r)*N + col] = acc[q][r] + bv;
  }
}

// ---------------- elementwise / reduction kernels ----------------
__global__ void k_zero_f(float* p, size_t n){
  for (size_t i = (size_t)blockIdx.x*blockDim.x + threadIdx.x; i < n; i += (size_t)gridDim.x*blockDim.x) p[i] = 0.f;
}
__global__ void k_zero_h(h16* p, size_t n){
  for (size_t i = (size_t)blockIdx.x*blockDim.x + threadIdx.x; i < n; i += (size_t)gridDim.x*blockDim.x) p[i] = (h16)0.f;
}
__global__ void k_embed(const int* z, const float* amap, float* x, int N){
  int idx = blockIdx.x*blockDim.x + threadIdx.x;
  if (idx >= N*9) return;
  int n = idx/9, c = idx%9;
  x[idx] = amap[z[n]*9 + c];
}
__global__ void k_gate(const float* G, int ldg, float* o32, h16* o16, int rows){
  int idx = blockIdx.x*blockDim.x + threadIdx.x;
  if (idx >= rows*256) return;
  int r = idx >> 8, c = idx & 255;
  float v = gate_val(G + (size_t)r*ldg, c);
  if (o32) o32[idx] = v;
  if (o16) o16[idx] = (h16)v;
}
__global__ void k_agg(const float* M2, const int* ei, float* agg, int e0, int rows, int E){
  int idx = blockIdx.x*blockDim.x + threadIdx.x;
  if (idx >= rows*256) return;
  int r = idx >> 8, c = idx & 255;
  int e = e0 + r;
  if (e >= E) return;
  atomicAdd(&agg[(size_t)ei[E+e]*256 + c], M2[idx]);
}
__global__ void k_cnt_edges(const int* ei, float* cnt, int E){
  int e = blockIdx.x*blockDim.x + threadIdx.x;
  if (e < E) atomicAdd(&cnt[ei[E+e]], 1.f);
}
__global__ void k_build_u(const float* x, int din, const float* agg, const float* cnt,
                          h16* Au, int Kpad, int N){
  size_t idx = (size_t)blockIdx.x*blockDim.x + threadIdx.x;
  if (idx >= (size_t)N*Kpad) return;
  int n = (int)(idx / Kpad), c = (int)(idx % Kpad);
  float v = 0.f;
  if (c < din) v = x[(size_t)n*din + c];
  else if (c < din + 256) v = agg[(size_t)n*256 + (c - din)] / fmaxf(cnt[n], 1.f);
  Au[idx] = (h16)v;
}
__global__ void k_cast16(const float* x, h16* o, int N){
  int idx = blockIdx.x*blockDim.x + threadIdx.x;
  if (idx < N*256) o[idx] = (h16)x[idx];
}
__global__ void k_bn_stats(const float* h, float* mean, float* msq, int N){
  int ch = blockIdx.x;
  int off, nc;
  if (ch < 54){ off = ch; nc = 1; }
  else if (ch < 71){ off = 54 + (ch-54)*3; nc = 3; }
  else if (ch < 88){ off = 105 + (ch-71)*3; nc = 3; }
  else if (ch < 98){ off = 156 + (ch-88)*5; nc = 5; }
  else { off = 206 + (ch-98)*5; nc = 5; }
  __shared__ float ss[256], sq[256];
  float a = 0.f, b = 0.f;
  for (int i = threadIdx.x; i < N*nc; i += blockDim.x){
    int n = i / nc, cc = i % nc;
    float f = h[(size_t)n*256 + off + cc];
    a += f; b += f*f;
  }
  ss[threadIdx.x] = a; sq[threadIdx.x] = b;
  __syncthreads();
  for (int s = 128; s > 0; s >>= 1){
    if (threadIdx.x < s){ ss[threadIdx.x] += ss[threadIdx.x+s]; sq[threadIdx.x] += sq[threadIdx.x+s]; }
    __syncthreads();
  }
  if (threadIdx.x == 0){
    mean[ch] = (ch < 54) ? ss[0]/(float)N : 0.f;
    msq[ch]  = sq[0]/(float)(N*nc);
  }
}
__global__ void k_bn_apply(const float* h, const float* mean, const float* msq,
                           const float* bw, const float* bb, float* xo, int N){
  int idx = blockIdx.x*blockDim.x + threadIdx.x;
  if (idx >= N*256) return;
  int c = idx & 255;
  float f = h[idx];
  if (c < 54){
    float mu = mean[c];
    float var = msq[c] - mu*mu;
    f = (f - mu)*rsqrtf(var + 1e-5f)*bw[c] + bb[c];
  } else {
    int ch;
    if (c < 105) ch = 54 + (c-54)/3;
    else if (c < 156) ch = 71 + (c-105)/3;
    else if (c < 206) ch = 88 + (c-156)/5;
    else ch = 98 + (c-206)/5;
    f = f*rsqrtf(msq[ch] + 1e-5f)*bw[ch];
  }
  xo[idx] = f;
}
__global__ void k_pool(const float* h, const int* batch, float* xg, int N){
  int idx = blockIdx.x*blockDim.x + threadIdx.x;
  if (idx >= N*256) return;
  int n = idx >> 8, c = idx & 255;
  atomicAdd(&xg[(size_t)batch[n]*256 + c], h[idx]);
}
__global__ void k_cntg(const int* batch, float* cg_, int N){
  int n = blockIdx.x*blockDim.x + threadIdx.x;
  if (n < N) atomicAdd(&cg_[batch[n]], 1.f);
}
__global__ void k_div64(const float* xg, const float* cg_, h16* A, int G){
  int idx = blockIdx.x*blockDim.x + threadIdx.x;
  if (idx < G*256) A[idx] = (h16)(xg[idx] / fmaxf(cg_[idx >> 8], 1.f));
}
__global__ void k_post2(const float* m64, const float* w, const float* b, float* out, int G){
  int g = threadIdx.x;
  if (g >= G) return;
  float acc = b[0];
  float s = rsqrtf(54.f);
  for (int u = 0; u < 54; ++u) acc += m64[g*256 + u]*w[u]*s;
  out[g] = acc;
}

// ---------------- host-side irreps / path builders ----------------
struct HIr { int m, l, p; };
static const HIr H_HID[5] = {{54,0,1},{17,1,1},{17,1,-1},{10,2,1},{10,2,-1}};
static const HIr H_GI[6]  = {{54,0,1},{54,0,1},{17,1,1},{17,1,-1},{10,2,1},{10,2,-1}};
static const HIr H_SH[3]  = {{1,0,1},{1,1,-1},{1,2,1}};

static int cgoff(int l1,int l2,int l3){
  for (int i=0;i<15;++i) if (H_L1[i]==l1 && H_L2[i]==l2 && H_L3[i]==l3) return H_OFF[i];
  return 0;
}
static inline int iabs_(int x){ return x<0? -x : x; }

static void build_msg(const HIr* base, int nbase, int din, void* const* d_in, int& pi, MsgDesc& D){
  HIr irm[11]; int nm = 0;
  for (int r = 0; r < 2; ++r) for (int i = 0; i < nbase; ++i) irm[nm++] = base[i];
  HIr sc; sc.m = 1; sc.l = 0; sc.p = 1; irm[nm++] = sc;
  int ioff[12]; { int o=0; for (int i=0;i<nm;++i){ ioff[i]=o; o += irm[i].m*(2*irm[i].l+1); } }
  int fan[6] = {0,0,0,0,0,0};
  for (int k=0;k<6;++k) for (int i=0;i<nm;++i) for (int j=0;j<3;++j){
    int l1=irm[i].l, l2=H_SH[j].l, l3=H_GI[k].l;
    if (irm[i].p*H_SH[j].p == H_GI[k].p && iabs_(l1-l2) <= l3 && l3 <= l1+l2) fan[k] += irm[i].m;
  }
  D.npaths = 0; int tt = 0;
  for (int k=0;k<6;++k) for (int i=0;i<nm;++i) for (int j=0;j<3;++j){
    int l1=irm[i].l, l2=H_SH[j].l, l3=H_GI[k].l;
    if (!(irm[i].p*H_SH[j].p == H_GI[k].p && iabs_(l1-l2) <= l3 && l3 <= l1+l2)) continue;
    int q = D.npaths++;
    D.w[q] = (const float*)d_in[pi++];
    D.in_off[q] = ioff[i]; D.t_off[q] = tt;
    D.cg[q] = (short)cgoff(l1,l2,l3);
    D.m1v[q] = (char)irm[i].m; D.m3v[q] = (char)H_GI[k].m;
    D.n1v[q] = (char)(2*l1+1); D.n2v[q] = (char)(2*l2+1); D.n3v[q] = (char)(2*l3+1);
    D.kk[q] = (char)k;
    tt += irm[i].m*(2*l3+1);
  }
  for (int k=0;k<6;++k)
    D.kscale[k] = fan[k] ? sqrtf((float)(2*H_GI[k].l+1))/sqrtf((float)fan[k]) : 0.f;
  D.din = din; D.ttot = tt;
}

static void build_lin(const HIr* in, int nin, const HIr* out, int nout,
                      void* const* d_in, int& pi, LinDesc& D, int Kpad, int Npad){
  int ioff[12]; { int o=0; for (int i=0;i<nin;++i){ ioff[i]=o; o += in[i].m*(2*in[i].l+1); } }
  int ooff[8];  { int o=0; for (int k=0;k<nout;++k){ ooff[k]=o; o += out[k].m*(2*out[k].l+1); } }
  int fan[8];
  for (int k=0;k<nout;++k){ fan[k]=0; for (int i=0;i<nin;++i) if (in[i].l==out[k].l && in[i].p==out[k].p) fan[k] += in[i].m; }
  D.npaths = 0; D.etotal = 0;
  for (int k=0;k<nout;++k) for (int i=0;i<nin;++i){
    if (!(in[i].l==out[k].l && in[i].p==out[k].p)) continue;
    int q = D.npaths++;
    D.w[q] = (const float*)d_in[pi++];
    D.in_off[q] = ioff[i]; D.out_off[q] = ooff[k];
    D.m1v[q] = in[i].m; D.m3v[q] = out[k].m; D.n3v[q] = 2*out[k].l+1;
    D.scale[q] = 1.f/sqrtf((float)fan[k]);
    D.ebase[q] = D.etotal;
    D.etotal += in[i].m*out[k].m*D.n3v[q];
  }
  D.Kpad = Kpad; D.Npad = Npad;
}

static inline int NB(long long n, int t){ return (int)((n + t - 1)/t); }

// GEMM launch helper: waves = mtiles * (ntiles/4), 8 waves per block
static inline int GEMM_BLK(int M, int N){ return NB((long long)(M/16)*((N/16)/4), 8); }

// ---------------- driver ----------------
extern "C" void kernel_launch(void* const* d_in, const int* in_sizes, int n_in,
                              void* d_out, int out_size, void* d_ws, size_t ws_size,
                              hipStream_t stream){
  const int*   z    = (const int*)  d_in[0];
  const float* pos  = (const float*)d_in[1];
  const int*   ei   = (const int*)  d_in[2];
  const float* cell = (const float*)d_in[3];
  const int*   bat  = (const int*)  d_in[4];
  const float* amap = (const float*)d_in[6];
  const int N = in_sizes[0];
  const int E = in_sizes[2]/2;
  const int G = (out_size > 0) ? out_size : 64;

  // ---- parameter walking (make_params insertion order) ----
  int pi = 7;
  MsgDesc md[4]; LinDesc m2d[4], u1d[4], u2d[4];
  const float *m1b[4], *m2b[4], *u1b[4], *u2b[4], *bnw[4], *bnb[4];
  for (int l = 0; l < 4; ++l){
    if (l == 0){ HIr b0; b0.m=9; b0.l=0; b0.p=1; build_msg(&b0, 1, 9, d_in, pi, md[l]); }
    else build_msg(H_HID, 5, 256, d_in, pi, md[l]);
    m1b[l] = (const float*)d_in[pi++];
    build_lin(H_HID, 5, H_GI, 6, d_in, pi, m2d[l], 256, 320);
    m2b[l] = (const float*)d_in[pi++];
    HIr uin[10]; int nu, Kp;
    if (l == 0){ uin[0].m=9; uin[0].l=0; uin[0].p=1; for (int i=0;i<5;++i) uin[1+i]=H_HID[i]; nu=6; Kp=288; }
    else { for (int i=0;i<5;++i){ uin[i]=H_HID[i]; uin[5+i]=H_HID[i]; } nu=10; Kp=512; }
    build_lin(uin, nu, H_GI, 6, d_in, pi, u1d[l], Kp, 320);
    u1b[l] = (const float*)d_in[pi++];
    build_lin(H_HID, 5, H_HID, 5, d_in, pi, u2d[l], 256, 256);
    u2b[l] = (const float*)d_in[pi++];
    bnw[l] = (const float*)d_in[pi++];
    bnb[l] = (const float*)d_in[pi++];
  }
  LinDesc pre1d, pre2d, post1d;
  build_lin(H_HID, 5, H_GI, 6, d_in, pi, pre1d, 256, 320);
  const float* pre1b = (const float*)d_in[pi++];
  build_lin(H_HID, 5, H_HID, 5, d_in, pi, pre2d, 256, 256);
  const float* pre2b = (const float*)d_in[pi++];
  build_lin(H_HID, 5, H_GI, 6, d_in, pi, post1d, 256, 320);
  const float* post1b = (const float*)d_in[pi++];
  const float* post2w = (const float*)d_in[pi++];
  const float* post2b = (const float*)d_in[pi++];

  // ---- workspace carve ----
  char* wsb = (char*)d_ws;
  size_t off = 0;
  auto carve = [&](size_t bytes) -> void* {
    void* p = wsb + off;
    off = (off + bytes + 255) & ~(size_t)255;
    return p;
  };
  const int CH = EDGE_CHUNK;
  float* W3J  = (float*)carve(615*sizeof(float));
  float* Xa   = (float*)carve((size_t)N*256*sizeof(float));
  float* Xb   = (float*)carve((size_t)N*256*sizeof(float));
  h16*   Mh   = (h16*)  carve((size_t)CH*256*sizeof(h16));
  float* Ge   = (float*)carve((size_t)CH*320*sizeof(float));
  float* M2f  = (float*)carve((size_t)CH*256*sizeof(float));
  float* agg  = (float*)carve((size_t)N*256*sizeof(float));
  float* cnt  = (float*)carve((size_t)N*sizeof(float));
  h16*   Au   = (h16*)  carve((size_t)N*512*sizeof(h16));
  h16*   Au2  = (h16*)  carve((size_t)N*256*sizeof(h16));
  float* Gn   = (float*)carve((size_t)N*320*sizeof(float));
  float* Hb   = (float*)carve((size_t)N*256*sizeof(float));
  h16*   Bm2  = (h16*)  carve((size_t)256*320*sizeof(h16));
  h16*   Bu   = (h16*)  carve((size_t)512*320*sizeof(h16));
  h16*   Bu2  = (h16*)  carve((size_t)256*256*sizeof(h16));
  float* meanb= (float*)carve(108*sizeof(float));
  float* msqb = (float*)carve(108*sizeof(float));
  float* xg   = (float*)carve((size_t)G*256*sizeof(float));
  float* cntg = (float*)carve((size_t)G*sizeof(float));
  h16*   A64h = (h16*)  carve((size_t)G*256*sizeof(h16));
  float* G64  = (float*)carve((size_t)G*320*sizeof(float));
  float* m64  = (float*)carve((size_t)G*256*sizeof(float));
  (void)ws_size; (void)n_in;

  // ---- forward ----
  k_init_w3j<<<1, 32, 0, stream>>>(W3J);
  k_embed<<<NB((long long)N*9,256),256,0,stream>>>(z, amap, Xa, N);
  k_zero_f<<<256,256,0,stream>>>(cnt, (size_t)N);
  k_cnt_edges<<<NB(E,256),256,0,stream>>>(ei, cnt, E);

  float* xc = Xa; float* xn = Xb; int din = 9;
  for (int l = 0; l < 4; ++l){
    k_zero_h<<<256,256,0,stream>>>(Bm2, (size_t)256*320);
    k_densify<<<NB(m2d[l].etotal,256),256,0,stream>>>(m2d[l], Bm2);
    k_zero_f<<<2048,256,0,stream>>>(agg, (size_t)N*256);
    for (int e0 = 0; e0 < E; e0 += CH){
      int rows = (E - e0 < CH) ? (E - e0) : CH;
      int Mr = (rows + 15) & ~15;
      k_msg<<<rows, 64, 0, stream>>>(md[l], xc, pos, ei, cell, W3J, m1b[l], Mh, e0, E);
      k_gemm<<<GEMM_BLK(Mr,320),256,0,stream>>>(Mh, Bm2, m2b[l], 54, Ge, Mr, 320, 256);
      k_gate<<<NB((long long)rows*256,256),256,0,stream>>>(Ge, 320, M2f, (h16*)nullptr, rows);
      k_agg<<<NB((long long)rows*256,256),256,0,stream>>>(M2f, ei, agg, e0, rows, E);
    }
    int Kp = (l == 0) ? 288 : 512;
    k_build_u<<<NB((long long)N*Kp,256),256,0,stream>>>(xc, din, agg, cnt, Au, Kp, N);
    k_zero_h<<<256,256,0,stream>>>(Bu, (size_t)Kp*320);
    k_densify<<<NB(u1d[l].etotal,256),256,0,stream>>>(u1d[l], Bu);
    k_gemm<<<GEMM_BLK(N,320),256,0,stream>>>(Au, Bu, u1b[l], 54, Gn, N, 320, Kp);
    k_gate<<<NB((long long)N*256,256),256,0,stream>>>(Gn, 320, (float*)nullptr, Au2, N);
    k_zero_h<<<256,256,0,stream>>>(Bu2, (size_t)256*256);
    k_densify<<<NB(u2d[l].etotal,256),256,0,stream>>>(u2d[l], Bu2);
    k_gemm<<<GEMM_BLK(N,256),256,0,stream>>>(Au2, Bu2, u2b[l], 54, Hb, N, 256, 256);
    k_bn_stats<<<108,256,0,stream>>>(Hb, meanb, msqb, N);
    k_bn_apply<<<NB((long long)N*256,256),256,0,stream>>>(Hb, meanb, msqb, bnw[l], bnb[l], xn, N);
    { float* t = xc; xc = xn; xn = t; }
    din = 256;
  }

  // head
  k_cast16<<<NB((long long)N*256,256),256,0,stream>>>(xc, Au2, N);
  k_zero_h<<<256,256,0,stream>>>(Bm2, (size_t)256*320);
  k_densify<<<NB(pre1d.etotal,256),256,0,stream>>>(pre1d, Bm2);
  k_gemm<<<GEMM_BLK(N,320),256,0,stream>>>(Au2, Bm2, pre1b, 54, Gn, N, 320, 256);
  k_gate<<<NB((long long)N*256,256),256,0,stream>>>(Gn, 320, (float*)nullptr, Au, N); // Au reused as Nx256 f16
  k_zero_h<<<256,256,0,stream>>>(Bu2, (size_t)256*256);
  k_densify<<<NB(pre2d.etotal,256),256,0,stream>>>(pre2d, Bu2);
  k_gemm<<<GEMM_BLK(N,256),256,0,stream>>>(Au, Bu2, pre2b, 54, Hb, N, 256, 256);
  k_zero_f<<<64,256,0,stream>>>(xg, (size_t)G*256);
  k_zero_f<<<1,64,0,stream>>>(cntg, (size_t)G);
  k_pool<<<NB((long long)N*256,256),256,0,stream>>>(Hb, bat, xg, N);
  k_cntg<<<NB(N,256),256,0,stream>>>(bat, cntg, N);
  k_div64<<<NB((long long)G*256,256),256,0,stream>>>(xg, cntg, A64h, G);
  k_zero_h<<<256,256,0,stream>>>(Bm2, (size_t)256*320);
  k_densify<<<NB(post1d.etotal,256),256,0,stream>>>(post1d, Bm2);
  k_gemm<<<GEMM_BLK(G,320),256,0,stream>>>(A64h, Bm2, post1b, 54, G64, G, 320, 256);
  k_gate<<<NB((long long)G*256,256),256,0,stream>>>(G64, 320, m64, (h16*)nullptr, G);
  k_post2<<<1, 64, 0, stream>>>(m64, post2w, post2b, (float*)d_out, G);
}
